// ScaledDotProductAttention_61598420959864
// MI455X (gfx1250) — matrix operand order repaired and checked by reading, not compiled, not run on hardware
//
#include <hip/hip_runtime.h>

// ---- CDNA5 WMMA vector types ----
typedef __bf16          v16bf __attribute__((ext_vector_type(16)));
typedef unsigned short  v16us __attribute__((ext_vector_type(16)));
typedef unsigned short  v8us  __attribute__((ext_vector_type(8)));
typedef float           v8f   __attribute__((ext_vector_type(8)));

#define B_    4
#define H_    16
#define S_    2048
#define D_    64
#define WAVES 8
#define BLOCK (WAVES * 32)
#define QW    16              // query rows per wave
#define QWG   (WAVES * QW)    // 128 query rows per workgroup
#define KB    32              // key block (WMMA K dim for bf16)
// 1/sqrt(64) * log2(e): softmax computed in exp2 domain (single v_exp_f32 each)
#define SCALE_L2E 0.1803368801111204f

union U16 { v16us u; v16bf v; };

__device__ __forceinline__ unsigned short f2bf(float f) {
    __bf16 h = (__bf16)f;                       // native fptrunc (RNE)
    return __builtin_bit_cast(unsigned short, h);
}

__device__ __forceinline__ v16bf asbf(v16us u) {
    U16 t; t.u = u; return t.v;
}

// xor-butterfly lane swap within 16-lane groups via a single ds_swizzle_b32
#define SWZ_XOR(x, m) __builtin_bit_cast(float, \
    __builtin_amdgcn_ds_swizzle(__builtin_bit_cast(int, (x)), ((m) << 10) | 0x1f))

__device__ __forceinline__ float rowmax16(float x) {
    x = fmaxf(x, SWZ_XOR(x, 1));
    x = fmaxf(x, SWZ_XOR(x, 2));
    x = fmaxf(x, SWZ_XOR(x, 4));
    x = fmaxf(x, SWZ_XOR(x, 8));
    return x;
}

// one thread's share of a K/V key-block held in VGPRs (loads-in-flight)
struct Blk { float4 ka, kb, va, vb; };

__device__ __forceinline__ Blk load_blk(const float* kp, const float* vp,
                                        int k0, int row, int c8) {
    Blk b;
    const float* ks = kp + (size_t)(k0 + row) * D_ + c8;
    b.ka = *(const float4*)(ks);
    b.kb = *(const float4*)(ks + 4);
    const float* vs = vp + (size_t)(k0 + row) * D_ + c8;
    b.va = *(const float4*)(vs);
    b.vb = *(const float4*)(vs + 4);
    return b;
}

__device__ __forceinline__ void store_blk(unsigned short* Ksb, unsigned short* Vtb,
                                          const Blk& b, int row, int c8) {
    v8us kp8;
    kp8[0] = f2bf(b.ka.x); kp8[1] = f2bf(b.ka.y);
    kp8[2] = f2bf(b.ka.z); kp8[3] = f2bf(b.ka.w);
    kp8[4] = f2bf(b.kb.x); kp8[5] = f2bf(b.kb.y);
    kp8[6] = f2bf(b.kb.z); kp8[7] = f2bf(b.kb.w);
    *(v8us*)(&Ksb[row * D_ + c8]) = kp8;
    // transpose V while staging: Vtb[d][key]
    Vtb[(c8 + 0) * KB + row] = f2bf(b.va.x);
    Vtb[(c8 + 1) * KB + row] = f2bf(b.va.y);
    Vtb[(c8 + 2) * KB + row] = f2bf(b.va.z);
    Vtb[(c8 + 3) * KB + row] = f2bf(b.va.w);
    Vtb[(c8 + 4) * KB + row] = f2bf(b.vb.x);
    Vtb[(c8 + 5) * KB + row] = f2bf(b.vb.y);
    Vtb[(c8 + 6) * KB + row] = f2bf(b.vb.z);
    Vtb[(c8 + 7) * KB + row] = f2bf(b.vb.w);
}

__global__ __launch_bounds__(BLOCK)
void fa_wmma_kernel(const float* __restrict__ q,
                    const float* __restrict__ k,
                    const float* __restrict__ v,
                    float* __restrict__ out)
{
    // double-buffered bf16 tiles: stage next block while matrix pipes chew current
    __shared__ __align__(32) unsigned short Ksb[2][KB * D_];        // 2 x 4 KB [key][d]
    __shared__ __align__(32) unsigned short Vtb[2][D_ * KB];        // 2 x 4 KB [d][key]
    __shared__ __align__(16) unsigned short Ppad[WAVES * QW * KB];  // 8 KB per-wave P scratch

    const int QBLKS = S_ / QWG;                 // 16
    const int bh    = blockIdx.x / QBLKS;
    const int qblk  = blockIdx.x % QBLKS;
    const int q0wg  = qblk * QWG;

    const int tid  = threadIdx.x;
    const int wave = tid >> 5;
    const int lane = tid & 31;
    const int half = lane >> 4;
    const int l16  = lane & 15;

    const float* qp = q   + (size_t)bh * S_ * D_;
    const float* kp = k   + (size_t)bh * S_ * D_;
    const float* vp = v   + (size_t)bh * S_ * D_;
    float*       op = out + (size_t)bh * S_ * D_;

    const int qw0 = q0wg + wave * QW;

    // ---- Q A-fragments (16x32 bf16 A-layout), rows qw0..qw0+15, two 32-wide k-chunks ----
    v16bf aq0, aq1;
    {
        const float* qrow = qp + (size_t)(qw0 + l16) * D_;
        U16 pk[2];
        #pragma unroll
        for (int c = 0; c < 2; ++c) {
            const int base = c * 32 + half * 16;
            const float4 x0 = *(const float4*)(qrow + base);
            const float4 x1 = *(const float4*)(qrow + base + 4);
            const float4 y0 = *(const float4*)(qrow + base + 8);
            const float4 y1 = *(const float4*)(qrow + base + 12);
            pk[c].u[0]  = f2bf(x0.x); pk[c].u[1]  = f2bf(x0.y);
            pk[c].u[2]  = f2bf(x0.z); pk[c].u[3]  = f2bf(x0.w);
            pk[c].u[4]  = f2bf(x1.x); pk[c].u[5]  = f2bf(x1.y);
            pk[c].u[6]  = f2bf(x1.z); pk[c].u[7]  = f2bf(x1.w);
            pk[c].u[8]  = f2bf(y0.x); pk[c].u[9]  = f2bf(y0.y);
            pk[c].u[10] = f2bf(y0.z); pk[c].u[11] = f2bf(y0.w);
            pk[c].u[12] = f2bf(y1.x); pk[c].u[13] = f2bf(y1.y);
            pk[c].u[14] = f2bf(y1.z); pk[c].u[15] = f2bf(y1.w);
        }
        aq0 = pk[0].v; aq1 = pk[1].v;
    }

    // all-ones 32x16 bf16 B fragment: every column of P*ones = row-sum, replicated
    v16bf ones;
    {
        U16 t;
        #pragma unroll
        for (int e = 0; e < 16; ++e) t.u[e] = 0x3F80;  // bf16(1.0)
        ones = t.v;
    }

    v8f o0 = {}, o1 = {}, o2 = {}, o3 = {};
    v8f Lf = {};                         // running softmax denominators (replicated/lane)
    float mrow[8];
    #pragma unroll
    for (int r = 0; r < 8; ++r) mrow[r] = -1e30f;

    const int nB = (q0wg + QWG) / KB;   // causal: only key blocks below this WG's diagonal

    const int row = tid >> 3;          // staged key row 0..31
    const int c8  = (tid & 7) * 8;     // staged d chunk of 8

    // ---- pipeline prologue: stage block 0 ----
    {
        Blk b0 = load_blk(kp, vp, 0, row, c8);
        store_blk(&Ksb[0][0], &Vtb[0][0], b0, row, c8);
    }
    __syncthreads();

    for (int kb = 0; kb < nB; ++kb) {
        const int k0 = kb * KB;
        const int p  = kb & 1;
        const bool have_next = (kb + 1 < nB);

        // issue next block's global loads NOW; latency overlaps this block's WMMAs
        Blk nxt;
        if (have_next) {
            nxt = load_blk(kp, vp, (kb + 1) * KB, row, c8);
            if (kb + 2 < nB) {  // warm L2 one block further out
                __builtin_prefetch(kp + (size_t)((kb + 2) * KB + row) * D_ + c8, 0, 0);
                __builtin_prefetch(vp + (size_t)((kb + 2) * KB + row) * D_ + c8, 0, 0);
            }
        }

        // wave-uniform causal skip (EXEC stays all-1s inside: WMMA-legal)
        if (k0 <= qw0 + QW - 1) {
            // ---- scores: S = Q K^T, two 16x16 tiles (keys 0..15 / 16..31) ----
            v8f s0, s1;
            #pragma unroll
            for (int t = 0; t < 2; ++t) {
                const int n = t * 16 + l16;          // key row within block
                const v16us b0 = *(const v16us*)(&Ksb[p][n * D_ + half * 16]);
                const v16us b1 = *(const v16us*)(&Ksb[p][n * D_ + half * 16 + 32]);
                v8f acc = {};
                acc = __builtin_amdgcn_wmma_f32_16x16x32_bf16(
                        false, aq0, false, asbf(b0), (short)0, acc, false, false);
                acc = __builtin_amdgcn_wmma_f32_16x16x32_bf16(
                        false, aq1, false, asbf(b1), (short)0, acc, false, false);
                if (t == 0) s0 = acc; else s1 = acc;
            }

            // ---- online softmax (exp2 domain; row = qw0 + half*8 + r) ----
            float p0v[8], p1v[8], alpha[8];
            #pragma unroll
            for (int r = 0; r < 8; ++r) {
                const int qrow = qw0 + half * 8 + r;
                float a0 = s0[r] * SCALE_L2E;
                float a1 = s1[r] * SCALE_L2E;
                if (k0 + l16      > qrow) a0 = -1e30f;   // causal mask
                if (k0 + 16 + l16 > qrow) a1 = -1e30f;
                const float mx   = rowmax16(fmaxf(a0, a1));   // ds_swizzle butterfly
                const float mnew = fmaxf(mrow[r], mx);
                alpha[r] = exp2f(mrow[r] - mnew);
                p0v[r]   = exp2f(a0 - mnew);
                p1v[r]   = exp2f(a1 - mnew);
                mrow[r]  = mnew;
            }
            #pragma unroll
            for (int r = 0; r < 8; ++r) {
                o0[r] *= alpha[r]; o1[r] *= alpha[r];
                o2[r] *= alpha[r]; o3[r] *= alpha[r];
                Lf[r] *= alpha[r];
            }

            // ---- P (C-layout) -> LDS -> A-layout bf16 fragment ----
            unsigned short* pw = &Ppad[wave * QW * KB];
            #pragma unroll
            for (int r = 0; r < 8; ++r) {
                const int prow = half * 8 + r;
                pw[prow * KB + l16]      = f2bf(p0v[r]);
                pw[prow * KB + 16 + l16] = f2bf(p1v[r]);
            }
            // same-wave DS ops are in-order; compiler inserts s_wait_dscnt
            const unsigned short* pr = &pw[l16 * KB];
            const v8us plo = *(const v8us*)(pr + half * 16);
            const v8us phi = *(const v8us*)(pr + half * 16 + 8);
            const v16us pu = __builtin_shufflevector(plo, phi,
                    0, 1, 2, 3, 4, 5, 6, 7, 8, 9, 10, 11, 12, 13, 14, 15);
            const v16bf pa = asbf(pu);

            // ---- denominators on the matrix pipe: L += P * ones ----
            Lf = __builtin_amdgcn_wmma_f32_16x16x32_bf16(
                    false, pa, false, ones, (short)0, Lf, false, false);

            // ---- O += P V : four 16-wide d-chunks, B from transposed bf16 V ----
            #pragma unroll
            for (int nc = 0; nc < 4; ++nc) {
                const v16us bv = *(const v16us*)(&Vtb[p][(nc * 16 + l16) * KB + half * 16]);
                v8f acc = (nc == 0) ? o0 : (nc == 1) ? o1 : (nc == 2) ? o2 : o3;
                acc = __builtin_amdgcn_wmma_f32_16x16x32_bf16(
                        false, pa, false, asbf(bv), (short)0, acc, false, false);
                if (nc == 0) o0 = acc; else if (nc == 1) o1 = acc;
                else if (nc == 2) o2 = acc; else o3 = acc;
            }
        }

        // ---- stage next block into the other buffer (after compute of current) ----
        if (have_next) {
            store_blk(&Ksb[p ^ 1][0], &Vtb[p ^ 1][0], nxt, row, c8);
        }
        __syncthreads();
    }

    // ---- epilogue: normalize (v_rcp_f32) and store ----
    #pragma unroll
    for (int r = 0; r < 8; ++r) {
        const float inv = __builtin_amdgcn_rcpf(Lf[r]);
        const size_t orow = (size_t)(qw0 + half * 8 + r) * D_;
        op[orow + 0 * 16 + l16] = o0[r] * inv;
        op[orow + 1 * 16 + l16] = o1[r] * inv;
        op[orow + 2 * 16 + l16] = o2[r] * inv;
        op[orow + 3 * 16 + l16] = o3[r] * inv;
    }
}

extern "C" void kernel_launch(void* const* d_in, const int* in_sizes, int n_in,
                              void* d_out, int out_size, void* d_ws, size_t ws_size,
                              hipStream_t stream) {
    (void)in_sizes; (void)n_in; (void)out_size; (void)d_ws; (void)ws_size;
    const float* q = (const float*)d_in[0];
    const float* k = (const float*)d_in[1];
    const float* v = (const float*)d_in[2];
    // d_in[3] is the causal mask; it's a deterministic tril, applied analytically.
    float* out = (float*)d_out;

    const int grid = B_ * H_ * (S_ / QWG);   // 4*16*16 = 1024 workgroups
    fa_wmma_kernel<<<grid, BLOCK, 0, stream>>>(q, k, v, out);
}
